// GraphAF_52175262711996
// MI455X (gfx1250) — matrix operand order repaired and checked by reading, not compiled, not run on hardware
//
#include <hip/hip_runtime.h>
#include <hip/hip_bf16.h>
#include <math.h>

// ---------------- problem constants ----------------
#define N_NODES 200000
#define BATCH   10000
#define N_EDGES 600000
#define C1      128
#define C2      256
#define NL1     256
#define NL2     512
#define EL1     256
#define EL2     512
#define EPS_F   1e-7f

typedef __attribute__((ext_vector_type(16))) __bf16 v16bf;
typedef __attribute__((ext_vector_type(8)))  float  v8f;

union Frag32B { uint4 q[2]; v16bf v; };

__device__ __forceinline__ unsigned short f2bf(float f) {
  union { float f; unsigned u; } x; x.f = f;
  unsigned r = x.u + 0x7FFFu + ((x.u >> 16) & 1u);  // round-to-nearest-even-ish
  return (unsigned short)(r >> 16);
}

// ---------------- utility kernels ----------------
__global__ void zero_kernel(float* __restrict__ p, long long n) {
  long long i = (long long)blockIdx.x * blockDim.x + threadIdx.x;
  long long stride = (long long)gridDim.x * blockDim.x;
  for (; i < n; i += stride) p[i] = 0.0f;
}

// scalar segment_sum scatter (used for the tiny IN=2 layer-1 messages)
__global__ void scatter_add_kernel(const float* __restrict__ src, int src_stride,
                                   float* __restrict__ dst, int dst_stride,
                                   const int* __restrict__ ei, int C, int nE) {
  long long idx = (long long)blockIdx.x * blockDim.x + threadIdx.x;
  if (idx >= (long long)nE * C) return;
  int e = (int)(idx / C);
  int c = (int)(idx - (long long)e * C);
  int s = ei[e];
  int d = ei[nE + e];
  atomicAdd(&dst[(size_t)d * dst_stride + c], src[(size_t)s * src_stride + c]);
}

// vectorized scatter: b128 load + 4 f32 atomics (dominant 0.6 GB edge pass)
__global__ void scatter_add4_kernel(const float* __restrict__ src, int src_stride,
                                    float* __restrict__ dst, int dst_stride,
                                    const int* __restrict__ ei, int C4, int nE) {
  long long idx = (long long)blockIdx.x * blockDim.x + threadIdx.x;
  if (idx >= (long long)nE * C4) return;
  int e  = (int)(idx / C4);
  int c4 = (int)(idx - (long long)e * C4);
  int s = ei[e];
  int d = ei[nE + e];
  const float4 v = *(const float4*)&src[(size_t)s * src_stride + c4 * 4];
  float* dp = &dst[(size_t)d * dst_stride + c4 * 4];
  atomicAdd(dp + 0, v.x);
  atomicAdd(dp + 1, v.y);
  atomicAdd(dp + 2, v.z);
  atomicAdd(dp + 3, v.w);
}

// layer 1: IN=2 so matmul degenerates to 2 FMAs per weight -> pure VALU.
// writes into columns [0,128) of Acat (row stride 256).
__global__ void conv1_kernel(const float* __restrict__ x, const float* __restrict__ msg,
                             const float* __restrict__ W1r, const float* __restrict__ W1n,
                             const float* __restrict__ b1, float* __restrict__ Acat) {
  long long idx = (long long)blockIdx.x * blockDim.x + threadIdx.x;
  if (idx >= (long long)N_NODES * C1) return;
  int n = (int)(idx >> 7);
  int c = (int)(idx & 127);
  float v = x[2 * n]     * W1r[2 * c]     + x[2 * n + 1]   * W1r[2 * c + 1]
          + msg[2 * n]   * W1n[2 * c]     + msg[2 * n + 1] * W1n[2 * c + 1]
          + b1[c];
  v = v > 0.0f ? v : 0.01f * v;  // leaky_relu 0.01
  Acat[(size_t)n * 256 + c] = v;
}

// Pack W (Cout x Ktot, row-major, optionally concatenation of two matrices along K)
// into bf16 WMMA B-fragments: frag(kt,ct) = 32 lanes x 8 dwords contiguous, frag
// index ft = kt*nct + ct. Contiguous-K 16-bit B layout (this code's order, same as the A fragments):
//   c = ct*16 + (lane&15),  k = kt*32 + (lane>>4)*16 + v*2 + j
__global__ void pack_w_kernel(const float* __restrict__ Wa, int strideA,
                              const float* __restrict__ Wb, int strideB,
                              int Ka, int Ktot, int Cout, unsigned* __restrict__ outp) {
  int idx = blockIdx.x * blockDim.x + threadIdx.x;  // output dword index
  int total = (Ktot >> 5) * (Cout >> 4) * 32 * 8;
  if (idx >= total) return;
  int v = idx & 7;
  int lane = (idx >> 3) & 31;
  int ft = idx >> 8;            // kt*nct + ct
  int nct = Cout >> 4;
  int kt = ft / nct;
  int ct = ft - kt * nct;
  int c = ct * 16 + (lane & 15);
  int k0 = kt * 32 + (lane >> 4) * 16 + v * 2;
  float f0 = (k0     < Ka) ? Wa[(size_t)c * strideA + k0]
                           : Wb[(size_t)c * strideB + (k0 - Ka)];
  float f1 = (k0 + 1 < Ka) ? Wa[(size_t)c * strideA + k0 + 1]
                           : Wb[(size_t)c * strideB + (k0 + 1 - Ka)];
  outp[idx] = (unsigned)f2bf(f0) | ((unsigned)f2bf(f1) << 16);
}

// ---------------- WMMA GEMM: C[M x Cout] = act(A[M x K] * Bpack + bias) ----------------
// block = 256 threads = 8 waves; block owns a 32-row tile staged fp32->bf16 in LDS
// (max 32x768 bf16 = 48 KB of the 320 KB WGP LDS). Each wave holds a 2x2 register
// tile (rows {0-15,16-31} x 2 column tiles) -> 4 v_wmma per K-step with A reused
// across columns and B reused across rows. Inner loop is branch-free; B fragments
// advance by pure pointer increment; next-K fragment prefetched branchlessly.
__global__ void __launch_bounds__(256)
gemm_wmma_kernel(const float* __restrict__ A, int lda, int K,
                 const unsigned* __restrict__ Bpack, int Cout,
                 const float* __restrict__ bias,
                 float* __restrict__ Cbuf, int ldc, int act, int M) {
  __shared__ __align__(16) unsigned short As[32 * 768];  // 48 KB worst case (K=768)

  const int r0 = blockIdx.x * 32;

  // stage A tile (fp32 -> bf16) into LDS; zero-pad rows past M
  for (int i = threadIdx.x; i < 32 * K; i += 256) {
    int row = i / K;
    int k = i - row * K;
    int gr = r0 + row;
    float f = (gr < M) ? A[(size_t)gr * lda + k] : 0.0f;
    As[i] = f2bf(f);
  }
  __syncthreads();

  const int lane = threadIdx.x & 31;   // wave32
  const int wave = threadIdx.x >> 5;
  const int m  = lane & 15;
  const int kh = lane >> 4;
  const int nkt = K >> 5;
  const int nct = Cout >> 4;
  const int fragStrideK = nct * 256;   // dwords between same-ct fragments, consecutive kt

  // A-fragment LDS element offsets (16-bit A 16x32 layout, matching B pack:
  // lanes 0-15 hold K {0-15}, lanes 16-31 hold K {16-31})
  const int aoff0 = m * K + kh * 16;          // row subtile 0 (rows 0-15)
  const int aoff1 = (m + 16) * K + kh * 16;   // row subtile 1 (rows 16-31)

  for (int ctp = wave * 2; ctp < nct; ctp += 16) {
    v8f acc00 = {}, acc01 = {}, acc10 = {}, acc11 = {};
    const unsigned* bp0 = Bpack + ((size_t)ctp * 32 + lane) * 8;
    const unsigned* bp1 = Bpack + ((size_t)(ctp + 1) * 32 + lane) * 8;

    for (int kt = 0; kt < nkt; ++kt) {
      Frag32B a0, a1, b0, b1;
      a0.q[0] = *(const uint4*)&As[aoff0 + kt * 32];
      a0.q[1] = *(const uint4*)&As[aoff0 + kt * 32 + 8];
      a1.q[0] = *(const uint4*)&As[aoff1 + kt * 32];
      a1.q[1] = *(const uint4*)&As[aoff1 + kt * 32 + 8];
      b0.q[0] = ((const uint4*)bp0)[0];
      b0.q[1] = ((const uint4*)bp0)[1];
      b1.q[0] = ((const uint4*)bp1)[0];
      b1.q[1] = ((const uint4*)bp1)[1];
      // branchless next-K prefetch (clamped on the last iteration)
      const unsigned* nxt = bp0 + ((kt + 1 < nkt) ? fragStrideK : 0);
      __builtin_prefetch(nxt, 0, 1);
      acc00 = __builtin_amdgcn_wmma_f32_16x16x32_bf16(false, a0.v, false, b0.v,
                                                      (short)0, acc00, false, false);
      acc01 = __builtin_amdgcn_wmma_f32_16x16x32_bf16(false, a0.v, false, b1.v,
                                                      (short)0, acc01, false, false);
      acc10 = __builtin_amdgcn_wmma_f32_16x16x32_bf16(false, a1.v, false, b0.v,
                                                      (short)0, acc10, false, false);
      acc11 = __builtin_amdgcn_wmma_f32_16x16x32_bf16(false, a1.v, false, b1.v,
                                                      (short)0, acc11, false, false);
      bp0 += fragStrideK;
      bp1 += fragStrideK;
    }

    // D layout: VGPR r, lanes 0-15 -> (M=r, N=lane); lanes 16-31 -> (M=r+8, N=lane-16)
    const int col0 = ctp * 16 + m;
    const int col1 = col0 + 16;
    const float bb0 = bias[col0];
    const float bb1 = bias[col1];
#pragma unroll
    for (int r = 0; r < 8; ++r) {
      int rowA = r0 + r + kh * 8;        // row subtile 0
      int rowB = rowA + 16;              // row subtile 1
      float v00 = acc00[r] + bb0;
      float v01 = acc01[r] + bb1;
      float v10 = acc10[r] + bb0;
      float v11 = acc11[r] + bb1;
      if (act == 1) {
        v00 = v00 > 0.0f ? v00 : 0.01f * v00;
        v01 = v01 > 0.0f ? v01 : 0.01f * v01;
        v10 = v10 > 0.0f ? v10 : 0.01f * v10;
        v11 = v11 > 0.0f ? v11 : 0.01f * v11;
      } else if (act == 2) {
        v00 = tanhf(v00); v01 = tanhf(v01);
        v10 = tanhf(v10); v11 = tanhf(v11);
      }
      if (rowA < M) {
        Cbuf[(size_t)rowA * ldc + col0] = v00;
        Cbuf[(size_t)rowA * ldc + col1] = v01;
      }
      if (rowB < M) {
        Cbuf[(size_t)rowB * ldc + col0] = v10;
        Cbuf[(size_t)rowB * ldc + col1] = v11;
      }
    }
  }
}

// pooled readout: read_g[batch[n], c4*4 .. +3] += g2[n, ...] (b128 load + 4 atomics)
__global__ void readout4_kernel(const float* __restrict__ g2, const int* __restrict__ batch,
                                float* __restrict__ outbuf, int out_stride) {
  long long idx = (long long)blockIdx.x * blockDim.x + threadIdx.x;
  if (idx >= (long long)N_NODES * (C2 / 4)) return;
  int n  = (int)(idx >> 6);
  int c4 = (int)(idx & 63);
  const float4 v = *(const float4*)&g2[(size_t)n * C2 + c4 * 4];
  float* dp = &outbuf[(size_t)batch[n] * out_stride + c4 * 4];
  atomicAdd(dp + 0, v.x);
  atomicAdd(dp + 1, v.y);
  atomicAdd(dp + 2, v.z);
  atomicAdd(dp + 3, v.w);
}

// edge-MLP input columns [256,512)=g2[idi], [512,768)=g2[idj] (b128 copies)
__global__ void gather_edge_kernel(const float* __restrict__ g2,
                                   const int* __restrict__ idi, const int* __restrict__ idj,
                                   float* __restrict__ einA) {
  long long idx = (long long)blockIdx.x * blockDim.x + threadIdx.x;
  if (idx >= (long long)BATCH * (C2 / 4)) return;
  int b  = (int)(idx >> 6);
  int c4 = (int)(idx & 63);
  float4 vi = *(const float4*)&g2[(size_t)idi[b] * 256 + c4 * 4];
  float4 vj = *(const float4*)&g2[(size_t)idj[b] * 256 + c4 * 4];
  *(float4*)&einA[(size_t)b * 768 + 256 + c4 * 4] = vi;
  *(float4*)&einA[(size_t)b * 768 + 512 + c4 * 4] = vj;
}

// final 512->4 head + NLL loss (x graph)
__global__ void loss_x_kernel(const float* __restrict__ H, const float* __restrict__ W3,
                              const float* __restrict__ b3, const float* __restrict__ y,
                              float* __restrict__ out) {
  int b = blockIdx.x * blockDim.x + threadIdx.x;
  if (b >= BATCH) return;
  float acc[4] = { b3[0], b3[1], b3[2], b3[3] };
  const float* h = H + (size_t)b * NL2;
  for (int k = 0; k < NL2; ++k) {
    float hv = h[k];
    acc[0] += hv * W3[0 * NL2 + k];
    acc[1] += hv * W3[1 * NL2 + k];
    acc[2] += hv * W3[2 * NL2 + k];
    acc[3] += hv * W3[3 * NL2 + k];
  }
#pragma unroll
  for (int j = 0; j < 2; ++j) {
    float a  = acc[j] * acc[j] + EPS_F;
    float mu = acc[2 + j];
    float e  = (y[b * 2 + j] - mu) / a;
    out[b * 2 + j] = e * e + logf(a);
  }
}

// final 512->2 head + NLL loss (A graph)
__global__ void loss_A_kernel(const float* __restrict__ H, const float* __restrict__ W3,
                              const float* __restrict__ b3, const float* __restrict__ y,
                              float* __restrict__ out) {
  int b = blockIdx.x * blockDim.x + threadIdx.x;
  if (b >= BATCH) return;
  float a0 = b3[0], a1 = b3[1];
  const float* h = H + (size_t)b * EL2;
  for (int k = 0; k < EL2; ++k) {
    float hv = h[k];
    a0 += hv * W3[0 * EL2 + k];
    a1 += hv * W3[1 * EL2 + k];
  }
  float a  = a0 * a0 + EPS_F;
  float e  = (y[b] - a1) / a;
  out[b] = e * e + logf(a);
}

// ---------------- host orchestration ----------------
static inline int gblk(long long n, int t) {
  long long b = (n + t - 1) / t;
  return (int)b;
}

extern "C" void kernel_launch(void* const* d_in, const int* in_sizes, int n_in,
                              void* d_out, int out_size, void* d_ws, size_t ws_size,
                              hipStream_t stream) {
  const float* x_x     = (const float*)d_in[0];
  const float* y_x     = (const float*)d_in[1];
  const int*   ei_x    = (const int*)  d_in[2];
  const int*   batch_x = (const int*)  d_in[3];
  const float* x_A     = (const float*)d_in[4];
  const float* y_A     = (const float*)d_in[5];
  const int*   ei_A    = (const int*)  d_in[6];
  const int*   batch_A = (const int*)  d_in[7];
  const int*   idi_A   = (const int*)  d_in[8];
  const int*   idj_A   = (const int*)  d_in[9];
  const float* W1r = (const float*)d_in[10];
  const float* W1n = (const float*)d_in[11];
  const float* b1  = (const float*)d_in[12];
  const float* W2r = (const float*)d_in[13];
  const float* W2n = (const float*)d_in[14];
  const float* b2  = (const float*)d_in[15];
  const float* nW1 = (const float*)d_in[16];
  const float* nb1 = (const float*)d_in[17];
  const float* nW2 = (const float*)d_in[18];
  const float* nb2 = (const float*)d_in[19];
  const float* nW3 = (const float*)d_in[20];
  const float* nb3 = (const float*)d_in[21];
  const float* eW1 = (const float*)d_in[22];
  const float* eb1 = (const float*)d_in[23];
  const float* eW2 = (const float*)d_in[24];
  const float* eb2 = (const float*)d_in[25];
  const float* eW3 = (const float*)d_in[26];
  const float* eb3 = (const float*)d_in[27];
  float* out = (float*)d_out;

  // workspace layout (floats)
  float* ws = (float*)d_ws;
  size_t off = 0;
  float* msg1  = ws + off; off += (size_t)N_NODES * 2;
  float* Acat  = ws + off; off += (size_t)N_NODES * 256;   // [g1 | msg2]
  float* g2    = ws + off; off += (size_t)N_NODES * 256;
  float* readx = ws + off; off += (size_t)BATCH * 256;
  float* einA  = ws + off; off += (size_t)BATCH * 768;
  float* h1    = ws + off; off += (size_t)BATCH * 256;
  float* h2    = ws + off; off += (size_t)BATCH * 512;
  unsigned* pWcat = (unsigned*)(ws + off);
  unsigned* pnW1  = pWcat + 256 * 256 / 2;
  unsigned* pnW2  = pnW1  + 256 * 256 / 2;
  unsigned* peW1  = pnW2  + 256 * 512 / 2;
  unsigned* peW2  = peW1  + 768 * 256 / 2;

  const int T = 256;

  // ---- pack all GEMM weights to bf16 WMMA fragment order (L2-resident) ----
  { int tot = 256 * 256 / 2; pack_w_kernel<<<gblk(tot, T), T, 0, stream>>>(W2r, 128, W2n, 128, 128, 256, 256, pWcat); }
  { int tot = 256 * 256 / 2; pack_w_kernel<<<gblk(tot, T), T, 0, stream>>>(nW1, 256, nW1, 256, 256, 256, 256, pnW1); }
  { int tot = 256 * 512 / 2; pack_w_kernel<<<gblk(tot, T), T, 0, stream>>>(nW2, 256, nW2, 256, 256, 256, 512, pnW2); }
  { int tot = 768 * 256 / 2; pack_w_kernel<<<gblk(tot, T), T, 0, stream>>>(eW1, 768, eW1, 768, 768, 768, 256, peW1); }
  { int tot = 256 * 512 / 2; pack_w_kernel<<<gblk(tot, T), T, 0, stream>>>(eW2, 256, eW2, 256, 256, 256, 512, peW2); }

  // ================= graph x backbone =================
  zero_kernel<<<4096, T, 0, stream>>>(msg1, (long long)N_NODES * 2);
  zero_kernel<<<65536, T, 0, stream>>>(Acat, (long long)N_NODES * 256);
  scatter_add_kernel<<<gblk((long long)N_EDGES * 2, T), T, 0, stream>>>(x_x, 2, msg1, 2, ei_x, 2, N_EDGES);
  conv1_kernel<<<gblk((long long)N_NODES * C1, T), T, 0, stream>>>(x_x, msg1, W1r, W1n, b1, Acat);
  scatter_add4_kernel<<<gblk((long long)N_EDGES * (C1 / 4), T), T, 0, stream>>>(Acat, 256, Acat + 128, 256, ei_x, C1 / 4, N_EDGES);
  gemm_wmma_kernel<<<(N_NODES + 31) / 32, T, 0, stream>>>(Acat, 256, 256, pWcat, 256, b2, g2, 256, /*leaky*/1, N_NODES);
  zero_kernel<<<4096, T, 0, stream>>>(readx, (long long)BATCH * 256);
  readout4_kernel<<<gblk((long long)N_NODES * (C2 / 4), T), T, 0, stream>>>(g2, batch_x, readx, 256);

  // ================= graph A backbone =================
  zero_kernel<<<4096, T, 0, stream>>>(msg1, (long long)N_NODES * 2);
  zero_kernel<<<65536, T, 0, stream>>>(Acat, (long long)N_NODES * 256);
  scatter_add_kernel<<<gblk((long long)N_EDGES * 2, T), T, 0, stream>>>(x_A, 2, msg1, 2, ei_A, 2, N_EDGES);
  conv1_kernel<<<gblk((long long)N_NODES * C1, T), T, 0, stream>>>(x_A, msg1, W1r, W1n, b1, Acat);
  scatter_add4_kernel<<<gblk((long long)N_EDGES * (C1 / 4), T), T, 0, stream>>>(Acat, 256, Acat + 128, 256, ei_A, C1 / 4, N_EDGES);
  gemm_wmma_kernel<<<(N_NODES + 31) / 32, T, 0, stream>>>(Acat, 256, 256, pWcat, 256, b2, g2, 256, /*leaky*/1, N_NODES);
  zero_kernel<<<8192, T, 0, stream>>>(einA, (long long)BATCH * 768);
  readout4_kernel<<<gblk((long long)N_NODES * (C2 / 4), T), T, 0, stream>>>(g2, batch_A, einA, 768);
  gather_edge_kernel<<<gblk((long long)BATCH * (C2 / 4), T), T, 0, stream>>>(g2, idi_A, idj_A, einA);

  // ================= node MLP (x) =================
  gemm_wmma_kernel<<<(BATCH + 31) / 32, T, 0, stream>>>(readx, 256, 256, pnW1, 256, nb1, h1, 256, /*tanh*/2, BATCH);
  gemm_wmma_kernel<<<(BATCH + 31) / 32, T, 0, stream>>>(h1,   256, 256, pnW2, 512, nb2, h2, 512, /*tanh*/2, BATCH);
  loss_x_kernel<<<gblk(BATCH, T), T, 0, stream>>>(h2, nW3, nb3, y_x, out);

  // ================= edge MLP (A) =================
  gemm_wmma_kernel<<<(BATCH + 31) / 32, T, 0, stream>>>(einA, 768, 768, peW1, 256, eb1, h1, 256, /*tanh*/2, BATCH);
  gemm_wmma_kernel<<<(BATCH + 31) / 32, T, 0, stream>>>(h1,  256, 256, peW2, 512, eb2, h2, 512, /*tanh*/2, BATCH);
  loss_A_kernel<<<gblk(BATCH, T), T, 0, stream>>>(h2, eW3, eb3, y_A, out + 2 * BATCH);

  (void)in_sizes; (void)n_in; (void)out_size; (void)ws_size;
}
